// CrossAttentionDecoder_74105365725826
// MI455X (gfx1250) — compile-verified, matrix operand order checked by reading, not run on hardware
//
#include <hip/hip_runtime.h>
#include <hip/hip_bf16.h>

typedef __attribute__((ext_vector_type(16))) __bf16 v16bf;
typedef __attribute__((ext_vector_type(4)))  __bf16 v4bf;
typedef __attribute__((ext_vector_type(8)))  float  v8f;

#define DIM    512
#define NLAT   128
#define NTOK   16
#define NHEAD  8
#define DH     64
#define NBATCH 1024

// Padded LDS row strides (elements): conflict-free banks, 16B-aligned fetches.
#define SP   520   // 512-wide bf16 matrices (q, x/out, l-tile, k-tile)
#define VTS  136   // V-transposed (512 x 128) bf16
#define SCS  132   // scores (128 x 128) f32
#define PS   136   // probabilities (128 x 128) bf16

// LDS byte offsets
#define OFF_Q   0u
#define OFF_XO  16640u          // 16*SP*2
#define OFF_LT  33280u
#define OFF_KT  49920u
#define OFF_VT  66560u          // 512*VTS*2 = 139264
#define OFF_SC  205824u         // 128*SCS*4 = 67584
#define OFF_P   273408u         // 128*PS*2  = 34816
#define SMEM_BYTES 308224u

// ---- WMMA operand loader (wave32, documented gfx1250 layouts) --------------
// A: 16x32 bf16 from row-major [rows x stride]; B: 32x16 bf16 from N-major
// storage (element (k,n) at base[n*stride+k]). Same address pattern for both:
//   lane L: row/N = rn0 + L%16 ; K = k0 + 8*(L/16) (+16) -> two 16B loads.
__device__ __forceinline__ v16bf ld_ab(const __bf16* __restrict__ base,
                                       int rn0, int k0, int stride) {
  const int lane = threadIdx.x & 31;
  const int rn = rn0 + (lane & 15);
  const int kk = k0 + ((lane >> 4) << 3);
  union { v16bf v; uint4 q[2]; } u;
  u.q[0] = *(const uint4*)(base + rn * stride + kk);
  u.q[1] = *(const uint4*)(base + rn * stride + kk + 16);
  return u.v;
}

// One 16x16 tile: C += A(16 x 32*ksteps) * B(32*ksteps x 16)
__device__ __forceinline__ v8f mm_tile(const __bf16* __restrict__ A, int kA0, int sA,
                                       const __bf16* __restrict__ B, int n0, int kB0, int sB,
                                       int ksteps, v8f c) {
  for (int kt = 0; kt < ksteps; ++kt) {
    v16bf a = ld_ab(A, 0,  kA0 + 32 * kt, sA);
    v16bf b = ld_ab(B, n0, kB0 + 32 * kt, sB);
    c = __builtin_amdgcn_wmma_f32_16x16x32_bf16(false, a, false, b, (short)0, c,
                                                false, false);
  }
  return c;
}

// Two 16x16 tiles sharing the A operand: independent accumulation chains for
// the XDL pipeline, half the A-operand LDS traffic.
__device__ __forceinline__ void mm_tile2(const __bf16* __restrict__ A, int kA0, int sA,
                                         const __bf16* __restrict__ B, int n0a, int n0b,
                                         int kB0, int sB, int ksteps,
                                         v8f& ca, v8f& cb) {
  for (int kt = 0; kt < ksteps; ++kt) {
    v16bf a  = ld_ab(A, 0,   kA0 + 32 * kt, sA);
    v16bf b0 = ld_ab(B, n0a, kB0 + 32 * kt, sB);
    v16bf b1 = ld_ab(B, n0b, kB0 + 32 * kt, sB);
    ca = __builtin_amdgcn_wmma_f32_16x16x32_bf16(false, a, false, b0, (short)0, ca,
                                                 false, false);
    cb = __builtin_amdgcn_wmma_f32_16x16x32_bf16(false, a, false, b1, (short)0, cb,
                                                 false, false);
  }
}

// C/D layout: lane L: N = L%16 ; VGPR r -> M = 8*(L/16) + r
__device__ __forceinline__ void st_c_bf16(__bf16* dst, int m0, int n0, int stride,
                                          v8f c, float scale) {
  const int lane = threadIdx.x & 31;
  const int n  = n0 + (lane & 15);
  const int mb = m0 + ((lane >> 4) << 3);
#pragma unroll
  for (int r = 0; r < 8; ++r)
    dst[(mb + r) * stride + n] = (__bf16)(c[r] * scale);
}

// store transposed: element (M=j, N=d) -> dst[d*stride + j]
__device__ __forceinline__ void st_c_bf16_T(__bf16* dst, int j0, int d0, int stride,
                                            v8f c) {
  const int lane = threadIdx.x & 31;
  const int d  = d0 + (lane & 15);
  const int jb = j0 + ((lane >> 4) << 3);
#pragma unroll
  for (int r = 0; r < 8; ++r)
    dst[d * stride + jb + r] = (__bf16)c[r];
}

__device__ __forceinline__ void st_c_f32(float* dst, int m0, int n0, int stride, v8f c) {
  const int lane = threadIdx.x & 31;
  const int n  = n0 + (lane & 15);
  const int mb = m0 + ((lane >> 4) << 3);
#pragma unroll
  for (int r = 0; r < 8; ++r)
    dst[(mb + r) * stride + n] = c[r];
}

// Vectorized f32->bf16 staging: 16x512 global f32 -> padded-stride LDS bf16.
// float4 loads, packed 4xbf16 (8B) stores; groups never straddle a row.
__device__ __forceinline__ void stage_tile(const float* __restrict__ src,
                                           __bf16* __restrict__ dst, int tid) {
  const float4* s4 = (const float4*)src;
#pragma unroll
  for (int e = 0; e < 4; ++e) {
    int g = e * 512 + tid;               // group of 4 elems, 2048 groups total
    float4 f = s4[g];
    int idx = g * 4;
    v4bf h;
    h[0] = (__bf16)f.x; h[1] = (__bf16)f.y; h[2] = (__bf16)f.z; h[3] = (__bf16)f.w;
    *(v4bf*)(dst + (idx >> 9) * SP + (idx & 511)) = h;
  }
}

// ---- phase 0: f32 -> bf16 weight conversion --------------------------------
__global__ void convert_w_kernel(const float* __restrict__ wq, const float* __restrict__ wk,
                                 const float* __restrict__ wv, const float* __restrict__ wo,
                                 __bf16* __restrict__ dst) {
  const int W = DIM * DIM;
  int i = blockIdx.x * blockDim.x + threadIdx.x;
  for (int e = i; e < 4 * W; e += gridDim.x * blockDim.x) {
    const float* src = (e < W) ? wq : (e < 2 * W) ? wk : (e < 3 * W) ? wv : wo;
    dst[e] = (__bf16)src[e & (W - 1)];
  }
}

// ---- main kernel: one workgroup (16 waves) per batch element ---------------
__global__ void __launch_bounds__(512, 1)
cross_attn_kernel(const float* __restrict__ x, const float* __restrict__ l,
                  const __bf16* __restrict__ wq, const __bf16* __restrict__ wk,
                  const __bf16* __restrict__ wv, const __bf16* __restrict__ wo,
                  const float* __restrict__ bo, float* __restrict__ out) {
  extern __shared__ char smem[];
  __bf16* q     = (__bf16*)(smem + OFF_Q);    // 16 x 512 (stride SP), pre-scaled 1/8
  __bf16* xo    = (__bf16*)(smem + OFF_XO);   // x staging, later attn-out
  __bf16* ltile = (__bf16*)(smem + OFF_LT);   // 16 x 512 latent tile
  __bf16* ktile = (__bf16*)(smem + OFF_KT);   // 16 x 512 K tile
  __bf16* vt    = (__bf16*)(smem + OFF_VT);   // V transposed: 512 x 128 (stride VTS)
  float*  sc    = (float*)(smem + OFF_SC);    // scores: (8*16) x 128 (stride SCS)
  __bf16* p     = (__bf16*)(smem + OFF_P);    // probs:  (8*16) x 128 (stride PS)

  const int b    = blockIdx.x;
  const int tid  = threadIdx.x;
  const int wave = tid >> 5;
  const int r    = b & (NLAT - 1);                 // last allowed latent index
  const int ntv  = (((r >> 4) + 1) + 1) & ~1;      // live j-tiles, rounded up to even

  const float* xb = x + (size_t)b * NTOK * DIM;
  const float* lb = l + (size_t)b * NLAT * DIM;

  // stage x -> bf16 LDS
  stage_tile(xb, xo, tid);
  __syncthreads();

  // Q = x @ Wq^T, folded score scale dh^-0.5 = 1/8; wave owns cols {w, w+16}
  {
    v8f ca = {}, cb = {};
    mm_tile2(xo, 0, SP, wq, wave * 16, (wave + 16) * 16, 0, DIM, 16, ca, cb);
    st_c_bf16(q, 0, wave * 16, SP, ca, 0.125f);
    st_c_bf16(q, 0, (wave + 16) * 16, SP, cb, 0.125f);
  }
  __syncthreads();

  for (int jt = 0; jt < ntv; ++jt) {
    // stage 16 rows of l -> bf16 LDS; prefetch next tile
    const float* lt = lb + jt * NTOK * DIM;
    if (jt + 1 < ntv)
      __builtin_prefetch(lt + NTOK * DIM + tid * 16, 0, 1);
    stage_tile(lt, ltile, tid);
    __syncthreads();

    if (wave < 8) {        // K tile: column tiles {w, w+8} and {w+16, w+24}
      for (int t = wave; t < 16; t += 16) { }  // (kept trivial; explicit pairs below)
      {
        v8f ca = {}, cb = {};
        mm_tile2(ltile, 0, SP, wk, wave * 16, (wave + 8) * 16, 0, DIM, 16, ca, cb);
        st_c_bf16(ktile, 0, wave * 16, SP, ca, 1.0f);
        st_c_bf16(ktile, 0, (wave + 8) * 16, SP, cb, 1.0f);
      }
      {
        v8f ca = {}, cb = {};
        mm_tile2(ltile, 0, SP, wk, (wave + 16) * 16, (wave + 24) * 16, 0, DIM, 16, ca, cb);
        st_c_bf16(ktile, 0, (wave + 16) * 16, SP, ca, 1.0f);
        st_c_bf16(ktile, 0, (wave + 24) * 16, SP, cb, 1.0f);
      }
    } else {               // V tile, stored transposed (d-major)
      const int w = wave - 8;
      {
        v8f ca = {}, cb = {};
        mm_tile2(ltile, 0, SP, wv, w * 16, (w + 8) * 16, 0, DIM, 16, ca, cb);
        st_c_bf16_T(vt, 16 * jt, w * 16, VTS, ca);
        st_c_bf16_T(vt, 16 * jt, (w + 8) * 16, VTS, cb);
      }
      {
        v8f ca = {}, cb = {};
        mm_tile2(ltile, 0, SP, wv, (w + 16) * 16, (w + 24) * 16, 0, DIM, 16, ca, cb);
        st_c_bf16_T(vt, 16 * jt, (w + 16) * 16, VTS, ca);
        st_c_bf16_T(vt, 16 * jt, (w + 24) * 16, VTS, cb);
      }
    }
    __syncthreads();

    if (wave < 8) {        // scores for this j-tile: head = wave
      const int h = wave;
      v8f c = {};
      c = mm_tile(q, h * DH, SP, ktile, 0, h * DH, SP, 2, c);
      st_c_f32(sc, h * NTOK, 16 * jt, SCS, c);
    }
    __syncthreads();
  }

  // masked softmax: one thread per (head, query) row
  if (tid < NHEAD * NTOK) {
    const float* row = sc + tid * SCS;
    float m = -1e30f;
    for (int j = 0; j <= r; ++j) m = fmaxf(m, row[j]);
    float s = 0.f;
    for (int j = 0; j <= r; ++j) s += __expf(row[j] - m);
    const float inv = 1.0f / s;
    __bf16* pr = p + tid * PS;
    for (int j = 0; j <= r; ++j) pr[j] = (__bf16)(__expf(row[j] - m) * inv);
    for (int j = r + 1; j < ntv * 16; ++j) pr[j] = (__bf16)0.0f;  // pad to WMMA K
  }
  __syncthreads();

  // out = P @ V (per head) into xo. Wave w: head h = w/2, two d-col tiles of
  // the same head so the A operand (P rows of head h) is shared.
  {
    const int h  = wave >> 1;
    const int n0 = h * DH + (wave & 1) * 32;
    v8f ca = {}, cb = {};
    mm_tile2(p + h * NTOK * PS, 0, PS, vt, n0, n0 + 16, 0, VTS, ntv >> 1, ca, cb);
    st_c_bf16(xo, 0, n0, SP, ca, 1.0f);
    st_c_bf16(xo, 0, n0 + 16, SP, cb, 1.0f);
  }
  __syncthreads();

  // final = out @ Wo^T + bo -> global f32; wave owns cols {w, w+16}
  {
    float* ob = out + (size_t)b * NTOK * DIM;
    v8f ca = {}, cb = {};
    mm_tile2(xo, 0, SP, wo, wave * 16, (wave + 16) * 16, 0, DIM, 16, ca, cb);
    const int lane = tid & 31;
    const int mb = (lane >> 4) << 3;
    const int na = wave * 16 + (lane & 15);
    const int nb = (wave + 16) * 16 + (lane & 15);
    const float biasa = bo[na];
    const float biasb = bo[nb];
#pragma unroll
    for (int rr = 0; rr < 8; ++rr) {
      ob[(mb + rr) * DIM + na] = ca[rr] + biasa;
      ob[(mb + rr) * DIM + nb] = cb[rr] + biasb;
    }
  }
}

extern "C" void kernel_launch(void* const* d_in, const int* in_sizes, int n_in,
                              void* d_out, int out_size, void* d_ws, size_t ws_size,
                              hipStream_t stream) {
  const float* x  = (const float*)d_in[0];
  const float* l  = (const float*)d_in[1];
  const float* Wq = (const float*)d_in[2];
  const float* Wk = (const float*)d_in[3];
  const float* Wv = (const float*)d_in[4];
  const float* Wo = (const float*)d_in[5];
  const float* bo = (const float*)d_in[6];
  float* out = (float*)d_out;

  __bf16* wbf = (__bf16*)d_ws;               // 4 x 512*512 bf16 = 2 MB scratch
  const int W = DIM * DIM;

  convert_w_kernel<<<1024, 256, 0, stream>>>(Wq, Wk, Wv, Wo, wbf);

  (void)hipFuncSetAttribute((const void*)&cross_attn_kernel,
                            hipFuncAttributeMaxDynamicSharedMemorySize,
                            (int)SMEM_BYTES);
  cross_attn_kernel<<<NBATCH, 512, SMEM_BYTES, stream>>>(
      x, l, wbf, wbf + W, wbf + 2 * W, wbf + 3 * W, bo, out);
}
